// MoELayer_33011118637690
// MI455X (gfx1250) — matrix operand order repaired and checked by reading, not compiled, not run on hardware
//
#include <hip/hip_runtime.h>
#include <hip/hip_bf16.h>

// ---------------- types ----------------
typedef __attribute__((ext_vector_type(16))) __bf16 v16bf;
typedef __attribute__((ext_vector_type(8)))  __bf16 v8bf;
typedef __attribute__((ext_vector_type(8)))  float  v8f;

#define T_TOKENS   8192
#define DMODEL     2048
#define FFN_DIM    8192
#define HALF_FFN   4096
#define NEXPERTS   8
#define TOTAL_SLOTS (T_TOKENS * 2)

#define WMMA3(A, B, C) \
  __builtin_amdgcn_wmma_f32_16x16x32_bf16(false, (A), false, (B), (short)0, (C), false, false)

// ---------------- WMMA fragment loaders (CDNA5 16x16x32 bf16 layouts) ----------------
// A (16x32, MxK): lanes 0-15 hold M=lane, K pairs {0..7} then {16..23};
//                 lanes 16-31 hold M=lane-16, K pairs {8..15} then {24..31}.
__device__ __forceinline__ v16bf frag_a(const __bf16* row0, int ld, int lane) {
  const __bf16* p = row0 + (lane & 15) * ld + ((lane & 16) ? 8 : 0);
  v8bf lo = *(const v8bf*)(p);
  v8bf hi = *(const v8bf*)(p + 16);
  return __builtin_shufflevector(lo, hi, 0,1,2,3,4,5,6,7,8,9,10,11,12,13,14,15);
}
// B (32x16, KxN), stored in LDS as [n][k]: lanes 0-15 hold N=lane, K {0..7} then {16..23};
//                 lanes 16-31 hold N=lane-16, K {8..15} then {24..31}.
__device__ __forceinline__ v16bf frag_b(const __bf16* col0, int ld, int lane) {
  const __bf16* p = col0 + (lane & 15) * ld + ((lane & 16) ? 8 : 0);
  v8bf lo = *(const v8bf*)(p);
  v8bf hi = *(const v8bf*)(p + 16);
  return __builtin_shufflevector(lo, hi, 0,1,2,3,4,5,6,7,8,9,10,11,12,13,14,15);
}

__device__ __forceinline__ v8bf cvt8(float4 f0, float4 f1) {
  v8bf o;
  o[0] = (__bf16)f0.x; o[1] = (__bf16)f0.y; o[2] = (__bf16)f0.z; o[3] = (__bf16)f0.w;
  o[4] = (__bf16)f1.x; o[5] = (__bf16)f1.y; o[6] = (__bf16)f1.z; o[7] = (__bf16)f1.w;
  return o;
}

// ---------------- 1. zero output + counts ----------------
__global__ void moe_zero_kernel(float* __restrict__ out, int n, int* __restrict__ counts) {
  if (blockIdx.x == 0 && threadIdx.x < NEXPERTS) counts[threadIdx.x] = 0;
  int stride = gridDim.x * blockDim.x;
  for (int i = blockIdx.x * blockDim.x + threadIdx.x; i < n; i += stride) out[i] = 0.f;
}

// ---------------- 2. x fp32 -> bf16 ----------------
__global__ void moe_xcast_kernel(const float* __restrict__ x, __bf16* __restrict__ xbf) {
  size_t i = ((size_t)blockIdx.x * blockDim.x + threadIdx.x) * 8;
  float4 a = *(const float4*)(x + i);
  float4 b = *(const float4*)(x + i + 4);
  *(v8bf*)(xbf + i) = cvt8(a, b);
}

// ---------------- 3. gating: softmax + top-2 (one wave32 per token) ----------------
__global__ void moe_routing_kernel(const float* __restrict__ x, const float* __restrict__ gate_w,
                                   int* __restrict__ top_idx, float* __restrict__ top_w,
                                   int* __restrict__ counts) {
  int wid  = threadIdx.x >> 5;
  int lane = threadIdx.x & 31;
  int t = blockIdx.x * 8 + wid;
  if (t >= T_TOKENS) return;
  const float* xr = x + (size_t)t * DMODEL;
  float logits[NEXPERTS];
#pragma unroll
  for (int e = 0; e < NEXPERTS; ++e) {
    const float* g = gate_w + e * DMODEL;
    float s = 0.f;
    for (int j = lane; j < DMODEL; j += 32) s += xr[j] * g[j];
#pragma unroll
    for (int m = 16; m >= 1; m >>= 1) s += __shfl_xor(s, m, 32);
    logits[e] = s;
  }
  if (lane == 0) {
    int i0 = 0; float l0 = logits[0];
#pragma unroll
    for (int e = 1; e < NEXPERTS; ++e) if (logits[e] > l0) { l0 = logits[e]; i0 = e; }
    int i1 = (i0 == 0) ? 1 : 0; float l1 = logits[i1];
#pragma unroll
    for (int e = 0; e < NEXPERTS; ++e)
      if (e != i0 && e != i1 && logits[e] > l1) { l1 = logits[e]; i1 = e; }
    float w0 = 1.f / (1.f + __expf(l1 - l0));   // normalized top-2 softmax weight
    top_idx[t * 2 + 0] = i0; top_idx[t * 2 + 1] = i1;
    top_w [t * 2 + 0] = w0; top_w [t * 2 + 1] = 1.f - w0;
    atomicAdd(&counts[i0], 1);
    atomicAdd(&counts[i1], 1);
  }
}

// ---------------- 4. exclusive scan of expert counts ----------------
__global__ void moe_offsets_kernel(const int* __restrict__ counts, int* __restrict__ offsets,
                                   int* __restrict__ cursor) {
  if (blockIdx.x == 0 && threadIdx.x == 0) {
    int acc = 0;
#pragma unroll
    for (int e = 0; e < NEXPERTS; ++e) { offsets[e] = acc; acc += counts[e]; cursor[e] = 0; }
    offsets[NEXPERTS] = acc;
  }
}

// ---------------- 5. pack tokens into per-expert slot lists ----------------
__global__ void moe_scatter_kernel(const int* __restrict__ top_idx, const float* __restrict__ top_w,
                                   const int* __restrict__ offsets, int* __restrict__ cursor,
                                   int* __restrict__ token_of, float* __restrict__ coef) {
  int t = blockIdx.x * blockDim.x + threadIdx.x;
  if (t >= T_TOKENS) return;
#pragma unroll
  for (int k = 0; k < 2; ++k) {
    int e = top_idx[t * 2 + k];
    int s = offsets[e] + atomicAdd(&cursor[e], 1);
    token_of[s] = t;
    coef[s] = top_w[t * 2 + k];
  }
}

// ---------------- 6. up-projection GEMM + fused SwiGLU ----------------
// Block tile: 128 tokens x 64 'a'-columns, K-step 64 (two 32-K WMMA chunks per stage).
// 8 waves in a 4(M) x 2(N) grid; each wave owns a 32x32 output => 2x2 subtiles for both
// the gate half and the up half => 16 WMMAs per wave per stage.
__global__ void __launch_bounds__(256)
moe_up_kernel(const __bf16* __restrict__ xbf, const float* __restrict__ w_up,
              const int* __restrict__ counts, const int* __restrict__ offsets,
              const int* __restrict__ token_of, __bf16* __restrict__ aBuf) {
  const int e  = blockIdx.z;
  const int m0 = blockIdx.y * 128;
  const int nb = blockIdx.x * 64;               // column base within [0, HALF_FFN)
  const int ne = counts[e];
  if (m0 >= ne) return;
  const int nrem  = (ne - m0 < 128) ? (ne - m0) : 128;
  const int slot0 = offsets[e] + m0;

  __shared__ __align__(16) __bf16 As[128][72];
  __shared__ __align__(16) __bf16 Bs[2][64][72];

  const int tid  = threadIdx.x;
  const int lane = tid & 31;
  const int w    = tid >> 5;
  const int wm   = (w & 3) * 32;
  const int wn   = (w >> 2) * 32;

  // A staging: row = tid>>1 (0..127), 32 bf16 columns at (tid&1)*32
  const int ar = tid >> 1, ac = (tid & 1) * 32;
  const bool aval = (ar < nrem);
  const int tok = aval ? token_of[slot0 + ar] : 0;
  const __bf16* asrc = xbf + (size_t)tok * DMODEL + ac;

  // B staging: tile bt (0=gate rows, 1=up rows), row 0..63, 32 fp32 columns at (tid&1)*32
  const int bt = tid >> 7, brl = (tid & 127) >> 1, bc = (tid & 1) * 32;
  const float* bsrc = w_up + (size_t)e * FFN_DIM * DMODEL
                    + (size_t)(nb + brl + bt * HALF_FFN) * DMODEL + bc;

  v8f accg[2][2] = {};
  v8f accu[2][2] = {};

  for (int kb = 0; kb < DMODEL; kb += 64) {
    __syncthreads();
    {  // A: 64 bytes per thread
      uint4* d = (uint4*)&As[ar][ac];
      if (aval) {
        const uint4* s = (const uint4*)(asrc + kb);
        uint4 a0 = s[0], a1 = s[1], a2 = s[2], a3 = s[3];
        d[0] = a0; d[1] = a1; d[2] = a2; d[3] = a3;
      } else {
        uint4 z = {0u, 0u, 0u, 0u};
        d[0] = z; d[1] = z; d[2] = z; d[3] = z;
      }
    }
    {  // B: 32 fp32 per thread -> 32 bf16
      const float4* s = (const float4*)(bsrc + kb);
      __bf16* bd = &Bs[bt][brl][bc];
#pragma unroll
      for (int q = 0; q < 4; ++q)
        *(v8bf*)(bd + q * 8) = cvt8(s[2 * q], s[2 * q + 1]);
    }
    __syncthreads();

#pragma unroll
    for (int kk = 0; kk < 64; kk += 32) {
      v16bf a0 = frag_a(&As[wm][kk],      72, lane);
      v16bf a1 = frag_a(&As[wm + 16][kk], 72, lane);
      v16bf g0 = frag_b(&Bs[0][wn][kk],      72, lane);
      v16bf g1 = frag_b(&Bs[0][wn + 16][kk], 72, lane);
      v16bf u0 = frag_b(&Bs[1][wn][kk],      72, lane);
      v16bf u1 = frag_b(&Bs[1][wn + 16][kk], 72, lane);
      accg[0][0] = WMMA3(a0, g0, accg[0][0]);
      accg[0][1] = WMMA3(a0, g1, accg[0][1]);
      accg[1][0] = WMMA3(a1, g0, accg[1][0]);
      accg[1][1] = WMMA3(a1, g1, accg[1][1]);
      accu[0][0] = WMMA3(a0, u0, accu[0][0]);
      accu[0][1] = WMMA3(a0, u1, accu[0][1]);
      accu[1][0] = WMMA3(a1, u0, accu[1][0]);
      accu[1][1] = WMMA3(a1, u1, accu[1][1]);
    }
  }

  // epilogue: a = silu(g) * u -> bf16 activation buffer
  const int n  = lane & 15;
  const int mh = (lane & 16) ? 8 : 0;
#pragma unroll
  for (int mi = 0; mi < 2; ++mi)
#pragma unroll
    for (int ni = 0; ni < 2; ++ni)
#pragma unroll
      for (int v = 0; v < 8; ++v) {
        int m = wm + mi * 16 + v + mh;
        if (m < nrem) {
          float g = accg[mi][ni][v], u = accu[mi][ni][v];
          float a = (g / (1.f + __expf(-g))) * u;
          aBuf[(size_t)(slot0 + m) * HALF_FFN + nb + wn + ni * 16 + n] = (__bf16)a;
        }
      }
}

// ---------------- 7. down-projection GEMM + weighted scatter-add ----------------
// Block tile: 128 slots x 64 output columns, K-step 64; wave tile 32x32 => 8 WMMAs/stage.
__global__ void __launch_bounds__(256)
moe_down_kernel(const __bf16* __restrict__ aBuf, const float* __restrict__ w_down,
                const int* __restrict__ counts, const int* __restrict__ offsets,
                const int* __restrict__ token_of, const float* __restrict__ coef,
                float* __restrict__ out) {
  const int e  = blockIdx.z;
  const int m0 = blockIdx.y * 128;
  const int nb = blockIdx.x * 64;               // column base within [0, DMODEL)
  const int ne = counts[e];
  if (m0 >= ne) return;
  const int nrem  = (ne - m0 < 128) ? (ne - m0) : 128;
  const int slot0 = offsets[e] + m0;

  __shared__ __align__(16) __bf16 As[128][72];
  __shared__ __align__(16) __bf16 Bs[64][72];

  const int tid  = threadIdx.x;
  const int lane = tid & 31;
  const int w    = tid >> 5;
  const int wm   = (w & 3) * 32;
  const int wn   = (w >> 2) * 32;

  const int ar = tid >> 1, ac = (tid & 1) * 32;
  const bool aval = (ar < nrem);
  const __bf16* asrc = aBuf + (size_t)(slot0 + (aval ? ar : 0)) * HALF_FFN + ac;

  // B staging: row = tid>>2 (0..63), 16 fp32 columns at (tid&3)*16
  const int brl = tid >> 2, bc = (tid & 3) * 16;
  const float* bsrc = w_down + (size_t)e * DMODEL * HALF_FFN
                    + (size_t)(nb + brl) * HALF_FFN + bc;

  v8f acc[2][2] = {};

  for (int kb = 0; kb < HALF_FFN; kb += 64) {
    __syncthreads();
    {  // A: 64 bytes per thread
      uint4* d = (uint4*)&As[ar][ac];
      if (aval) {
        const uint4* s = (const uint4*)(asrc + kb);
        uint4 a0 = s[0], a1 = s[1], a2 = s[2], a3 = s[3];
        d[0] = a0; d[1] = a1; d[2] = a2; d[3] = a3;
      } else {
        uint4 z = {0u, 0u, 0u, 0u};
        d[0] = z; d[1] = z; d[2] = z; d[3] = z;
      }
    }
    {  // B: 16 fp32 per thread -> 16 bf16
      const float4* s = (const float4*)(bsrc + kb);
      __bf16* bd = &Bs[brl][bc];
      *(v8bf*)(bd)     = cvt8(s[0], s[1]);
      *(v8bf*)(bd + 8) = cvt8(s[2], s[3]);
    }
    __syncthreads();

#pragma unroll
    for (int kk = 0; kk < 64; kk += 32) {
      v16bf a0 = frag_a(&As[wm][kk],      72, lane);
      v16bf a1 = frag_a(&As[wm + 16][kk], 72, lane);
      v16bf b0 = frag_b(&Bs[wn][kk],      72, lane);
      v16bf b1 = frag_b(&Bs[wn + 16][kk], 72, lane);
      acc[0][0] = WMMA3(a0, b0, acc[0][0]);
      acc[0][1] = WMMA3(a0, b1, acc[0][1]);
      acc[1][0] = WMMA3(a1, b0, acc[1][0]);
      acc[1][1] = WMMA3(a1, b1, acc[1][1]);
    }
  }

  const int n  = lane & 15;
  const int mh = (lane & 16) ? 8 : 0;
#pragma unroll
  for (int mi = 0; mi < 2; ++mi)
#pragma unroll
    for (int ni = 0; ni < 2; ++ni)
#pragma unroll
      for (int v = 0; v < 8; ++v) {
        int m = wm + mi * 16 + v + mh;
        if (m < nrem) {
          int s = slot0 + m;
          int t = token_of[s];
          float c = coef[s];
          atomicAdd(out + (size_t)t * DMODEL + nb + wn + ni * 16 + n, c * acc[mi][ni][v]);
        }
      }
}

// ---------------- host launcher ----------------
extern "C" void kernel_launch(void* const* d_in, const int* in_sizes, int n_in,
                              void* d_out, int out_size, void* d_ws, size_t ws_size,
                              hipStream_t stream) {
  const float* x      = (const float*)d_in[0];
  const float* gate_w = (const float*)d_in[1];
  const float* w_up   = (const float*)d_in[2];
  const float* w_down = (const float*)d_in[3];
  float* out = (float*)d_out;

  char* ws = (char*)d_ws;
  size_t o = 0;
  __bf16* xbf     = (__bf16*)(ws + o); o += (size_t)T_TOKENS * DMODEL * 2;        // 33.5 MB
  __bf16* aBuf    = (__bf16*)(ws + o); o += (size_t)TOTAL_SLOTS * HALF_FFN * 2;   // 134 MB
  int*    top_idx = (int*)  (ws + o); o += (size_t)T_TOKENS * 2 * 4;
  float*  top_w   = (float*)(ws + o); o += (size_t)T_TOKENS * 2 * 4;
  int*    counts  = (int*)  (ws + o); o += 64;
  int*    cursor  = (int*)  (ws + o); o += 64;
  int*    offsets = (int*)  (ws + o); o += 64;
  int*    token_of= (int*)  (ws + o); o += (size_t)TOTAL_SLOTS * 4;
  float*  coef    = (float*)(ws + o); o += (size_t)TOTAL_SLOTS * 4;

  const int out_elems = T_TOKENS * DMODEL;
  moe_zero_kernel<<<4096, 256, 0, stream>>>(out, out_elems, counts);
  moe_xcast_kernel<<<(T_TOKENS * DMODEL / 8 + 255) / 256, 256, 0, stream>>>(x, xbf);
  moe_routing_kernel<<<T_TOKENS / 8, 256, 0, stream>>>(x, gate_w, top_idx, top_w, counts);
  moe_offsets_kernel<<<1, 32, 0, stream>>>(counts, offsets, cursor);
  moe_scatter_kernel<<<(T_TOKENS + 255) / 256, 256, 0, stream>>>(top_idx, top_w, offsets, cursor,
                                                                 token_of, coef);
  dim3 gu(HALF_FFN / 64, T_TOKENS / 128, NEXPERTS);
  moe_up_kernel<<<gu, 256, 0, stream>>>(xbf, w_up, counts, offsets, token_of, aBuf);
  dim3 gd(DMODEL / 64, T_TOKENS / 128, NEXPERTS);
  moe_down_kernel<<<gd, 256, 0, stream>>>(aBuf, w_down, counts, offsets, token_of, coef, out);
}
